// DecoderBlock_80985903333806
// MI455X (gfx1250) — matrix operand order repaired and checked by reading, not compiled, not run on hardware
//
#include <hip/hip_runtime.h>
#include <hip/hip_bf16.h>

typedef __bf16 bf16;
typedef __attribute__((ext_vector_type(8)))  bf16  bf16x8;
typedef __attribute__((ext_vector_type(16))) bf16  v16bf;
typedef __attribute__((ext_vector_type(8)))  float v8f;

#define DD  1024
#define SS  2048
#define MMK 2048
#define BBB 2
#define HHH 16
#define FFD 4096

// ---------------------------------------------------------------------------
// WMMA helpers (CDNA5 wave32, D = A(16x32 bf16) * B(32x16 bf16) + C(16x16 f32))
// ---------------------------------------------------------------------------
__device__ __forceinline__ v8f wmma_bf16(v16bf a, v16bf b, v8f c) {
  return __builtin_amdgcn_wmma_f32_16x16x32_bf16(false, a, false, b, (short)0, c,
                                                 false, false);
}

// B-fragment: 16 contiguous bf16 (lane n holds K contiguous)
__device__ __forceinline__ v16bf ld16(const bf16* p) {
  bf16x8 a = *(const bf16x8*)p;
  bf16x8 b = *(const bf16x8*)(p + 8);
  return __builtin_shufflevector(a, b, 0, 1, 2, 3, 4, 5, 6, 7, 8, 9, 10, 11, 12,
                                 13, 14, 15);
}
// A-fragment: {p[0..15]} contiguous K (matches B-fragment order)
__device__ __forceinline__ v16bf ldA(const bf16* p) {
  bf16x8 a = *(const bf16x8*)p;
  bf16x8 b = *(const bf16x8*)(p + 8);
  return __builtin_shufflevector(a, b, 0, 1, 2, 3, 4, 5, 6, 7, 8, 9, 10, 11, 12,
                                 13, 14, 15);
}

// Async global->LDS copy of 32 contiguous bytes (2 x b128).  The instruction
// offset is applied to BOTH the LDS and global addresses (ISA 10.x / 08_async),
// so one LDS-addr VGPR + one global-addr VGPR pair covers both beats.
__device__ __forceinline__ void async_cp32(unsigned lds_addr,
                                           const bf16* gptr) {
  asm volatile(
      "global_load_async_to_lds_b128 %0, %1, off\n\t"
      "global_load_async_to_lds_b128 %0, %1, off offset:16"
      :
      : "v"(lds_addr), "v"((unsigned long long)(uintptr_t)gptr)
      : "memory");
}
__device__ __forceinline__ void wait_async0() {
  asm volatile("s_wait_asynccnt 0x0" ::: "memory");
}

// ---------------------------------------------------------------------------
// fp32 -> bf16 converters (plain + transposing for weights)
// ---------------------------------------------------------------------------
__global__ void cvt_bf16_kernel(const float* __restrict__ s, bf16* __restrict__ d,
                                long n) {
  long i = (long)blockIdx.x * blockDim.x + threadIdx.x;
  if (i < n) d[i] = (bf16)s[i];
}

// src: [K][N] f32 row-major  ->  dst: [N][K] bf16 row-major (W^T)
__global__ void cvt_t_bf16_kernel(const float* __restrict__ s, bf16* __restrict__ d,
                                  int K, int N) {
  long i = (long)blockIdx.x * blockDim.x + threadIdx.x;
  if (i >= (long)K * N) return;
  int n = (int)(i / K);
  int k = (int)(i % K);
  d[i] = (bf16)s[(long)k * N + n];
}

// ---------------------------------------------------------------------------
// GEMM: C[M,N] = A[M,K](bf16) * Bt[N,K](bf16)^T + bias, optional ReLU,
// optional f32 and/or bf16 outputs.  Block = 256 thr (8 waves), tile 128x128x32.
// Staging uses async global->LDS copies tracked by ASYNCcnt.
// ---------------------------------------------------------------------------
__global__ __launch_bounds__(256) void gemm_bf16_kernel(
    const bf16* __restrict__ A, const bf16* __restrict__ Bt,
    const float* __restrict__ bias, float* __restrict__ outF,
    bf16* __restrict__ outB, int M, int N, int K, int relu) {
  __shared__ __align__(16) bf16 As[2][128][40];
  __shared__ __align__(16) bf16 Bs[2][128][40];

  const int tid = threadIdx.x;
  const int w = tid >> 5, l = tid & 31;
  const int ln = l & 15, hf = l >> 4;
  const int wm = w & 3, wn = w >> 2;  // 4 waves along M, 2 along N
  const int m0 = blockIdx.y * 128, n0 = blockIdx.x * 128;
  const int srow = tid >> 1;          // 0..127 staging row
  const int skc = (tid & 1) * 16;     // 0 or 16

  v8f acc[2][4] = {};

  const bf16* pa0 = A + (long)(m0 + srow) * K + skc;
  const bf16* pb0 = Bt + (long)(n0 + srow) * K + skc;
  const unsigned ldsA[2] = {(unsigned)(uintptr_t)&As[0][srow][skc],
                            (unsigned)(uintptr_t)&As[1][srow][skc]};
  const unsigned ldsB[2] = {(unsigned)(uintptr_t)&Bs[0][srow][skc],
                            (unsigned)(uintptr_t)&Bs[1][srow][skc]};

  auto stage = [&](int buf, int k0) {
    async_cp32(ldsA[buf], pa0 + k0);
    async_cp32(ldsB[buf], pb0 + k0);
  };

  stage(0, 0);
  wait_async0();
  __syncthreads();

  const int KT = K >> 5;
  for (int kt = 0; kt < KT; ++kt) {
    const int cur = kt & 1;
    if (kt + 2 < KT) {  // prefetch the tile after next (global_prefetch_b8)
      __builtin_prefetch(pa0 + (kt + 2) * 32, 0, 1);
      __builtin_prefetch(pb0 + (kt + 2) * 32, 0, 1);
    }
    if (kt + 1 < KT) stage(cur ^ 1, (kt + 1) * 32);

    v16bf af[2], bfr[4];
#pragma unroll
    for (int mi = 0; mi < 2; ++mi)
      af[mi] = ldA(&As[cur][wm * 32 + mi * 16 + ln][hf * 16]);
#pragma unroll
    for (int ni = 0; ni < 4; ++ni)
      bfr[ni] = ld16(&Bs[cur][wn * 64 + ni * 16 + ln][hf * 16]);
#pragma unroll
    for (int mi = 0; mi < 2; ++mi)
#pragma unroll
      for (int ni = 0; ni < 4; ++ni)
        acc[mi][ni] = wmma_bf16(af[mi], bfr[ni], acc[mi][ni]);

    wait_async0();      // next stage's LDS writes landed
    __syncthreads();
  }

#pragma unroll
  for (int mi = 0; mi < 2; ++mi) {
#pragma unroll
    for (int ni = 0; ni < 4; ++ni) {
      const int col = n0 + wn * 64 + ni * 16 + ln;
      const float bv = bias ? bias[col] : 0.f;
#pragma unroll
      for (int r = 0; r < 8; ++r) {
        const int row = m0 + wm * 32 + mi * 16 + r + hf * 8;
        float v = acc[mi][ni][r] + bv;
        if (relu) v = fmaxf(v, 0.f);
        if (outF) outF[(long)row * N + col] = v;
        if (outB) outB[(long)row * N + col] = (bf16)v;
      }
    }
  }
}

// ---------------------------------------------------------------------------
// Flash attention, hd=64, 64-key tiles, 4 waves x 16 query rows per block.
// Q/K/V are fp32 (projection outputs); converted to bf16 on staging.
// Out is bf16 [B, Sq, Dout] head-concatenated (feeds the next GEMM).
// ---------------------------------------------------------------------------
template <bool CAUSAL>
__global__ __launch_bounds__(128) void flash_attn_kernel(
    const float* __restrict__ Qb, const float* __restrict__ Kb,
    const float* __restrict__ Vb, bf16* __restrict__ Out, int Sq, int Sk,
    long qBS, long qSS, long kBS, long kSS, long vBS, long vSS, int Dout) {
  const int b = blockIdx.z, h = blockIdx.y, qt = blockIdx.x;
  const int tid = threadIdx.x;
  const int w = tid >> 5, l = tid & 31;
  const int ln = l & 15, hf = l >> 4;

  __shared__ __align__(16) bf16 Ks[64][72];        // [key][hd]
  __shared__ __align__(16) bf16 Vt[64][72];        // [hd][key]
  __shared__ __align__(16) bf16 Ps[4][16][72];     // per-wave P scratch

  const float* Q = Qb + (long)b * qBS + h * 64;
  const float* Kp = Kb + (long)b * kBS + h * 64;
  const float* Vp = Vb + (long)b * vBS + h * 64;

  const int qrow0 = qt * 64 + w * 16;

  // Q fragments (held for the whole kernel): 16 rows x 64 hd -> 2 A-frags
  v16bf qf[2];
  {
    const float* qr = Q + (long)(qrow0 + ln) * qSS;
#pragma unroll
    for (int ks = 0; ks < 2; ++ks) {
      const int c0 = ks * 32 + hf * 16;
      bf16x8 lo, hiv;
#pragma unroll
      for (int i = 0; i < 8; ++i) lo[i] = (bf16)qr[c0 + i];
#pragma unroll
      for (int i = 0; i < 8; ++i) hiv[i] = (bf16)qr[c0 + 8 + i];
      qf[ks] = __builtin_shufflevector(lo, hiv, 0, 1, 2, 3, 4, 5, 6, 7, 8, 9, 10,
                                       11, 12, 13, 14, 15);
    }
  }

  v8f oacc[4] = {};
  float mi[8], li[8];
#pragma unroll
  for (int r = 0; r < 8; ++r) { mi[r] = -3.0e38f; li[r] = 0.f; }

  const int nkt = CAUSAL ? (qt + 1) : (Sk >> 6);
  for (int kt = 0; kt < nkt; ++kt) {
    const int kb = kt * 64;
    {  // stage K (row-major) and V (transposed) tiles as bf16
      const int key = tid >> 1;
      const int c = (tid & 1) * 32;
      const float* kr = Kp + (long)(kb + key) * kSS + c;
      const float* vr = Vp + (long)(kb + key) * vSS + c;
#pragma unroll
      for (int i = 0; i < 32; ++i) Ks[key][c + i] = (bf16)kr[i];
#pragma unroll
      for (int i = 0; i < 32; ++i) Vt[c + i][key] = (bf16)vr[i];
    }
    __syncthreads();

    // S = Q K^T  (16x64 per wave; 4 N-tiles x 2 K-steps)
    v8f sacc[4] = {};
#pragma unroll
    for (int ni = 0; ni < 4; ++ni) {
#pragma unroll
      for (int ks = 0; ks < 2; ++ks) {
        v16bf kf = ld16(&Ks[ni * 16 + ln][ks * 32 + hf * 16]);
        sacc[ni] = wmma_bf16(qf[ks], kf, sacc[ni]);
      }
    }

    // online softmax per row slot
#pragma unroll
    for (int r = 0; r < 8; ++r) {
      const int row = qrow0 + r + hf * 8;
      float sv[4];
#pragma unroll
      for (int ni = 0; ni < 4; ++ni) {
        float s = sacc[ni][r] * 0.125f;  // 1/sqrt(64)
        if (CAUSAL) {
          const int key = kb + ni * 16 + ln;
          if (key > row) s = -3.0e38f;
        }
        sv[ni] = s;
      }
      float mx = fmaxf(fmaxf(sv[0], sv[1]), fmaxf(sv[2], sv[3]));
#pragma unroll
      for (int d = 1; d < 16; d <<= 1) mx = fmaxf(mx, __shfl_xor(mx, d, 32));
      const float mnew = fmaxf(mi[r], mx);
      const float corr = __expf(mi[r] - mnew);
      float psum = 0.f;
#pragma unroll
      for (int ni = 0; ni < 4; ++ni) {
        const float p = __expf(sv[ni] - mnew);
        psum += p;
        Ps[w][r + hf * 8][ni * 16 + ln] = (bf16)p;
      }
#pragma unroll
      for (int d = 1; d < 16; d <<= 1) psum += __shfl_xor(psum, d, 32);
      li[r] = li[r] * corr + psum;
      mi[r] = mnew;
#pragma unroll
      for (int ni = 0; ni < 4; ++ni) oacc[ni][r] *= corr;
    }

    // O += P @ V
#pragma unroll
    for (int ks = 0; ks < 2; ++ks) {
      v16bf pf = ldA(&Ps[w][ln][ks * 32 + hf * 16]);
#pragma unroll
      for (int ni = 0; ni < 4; ++ni) {
        v16bf vf = ld16(&Vt[ni * 16 + ln][ks * 32 + hf * 16]);
        oacc[ni] = wmma_bf16(pf, vf, oacc[ni]);
      }
    }
    __syncthreads();
  }

  // write bf16 head-concat output
#pragma unroll
  for (int ni = 0; ni < 4; ++ni) {
    const int col = h * 64 + ni * 16 + ln;
#pragma unroll
    for (int r = 0; r < 8; ++r) {
      const int row = qrow0 + r + hf * 8;
      const float v = oacc[ni][r] / li[r];
      Out[((long)b * Sq + row) * Dout + col] = (bf16)v;
    }
  }
}

// ---------------------------------------------------------------------------
// Fused residual add + LayerNorm; emits fp32 (residual) and optional bf16
// (next GEMM A operand).  One 256-thread block per row of D=1024.
// ---------------------------------------------------------------------------
__global__ __launch_bounds__(256) void add_ln_kernel(
    const float* __restrict__ X, const float* __restrict__ Y,
    const float* __restrict__ g, const float* __restrict__ bta,
    float* __restrict__ outF, bf16* __restrict__ outB, int D) {
  const int row = blockIdx.x;
  const int tid = threadIdx.x;
  __shared__ float red[256];
  __shared__ float red2[256];

  const float* xr = X + (long)row * D + tid * 4;
  const float* yr = Y + (long)row * D + tid * 4;
  float v[4];
  float s = 0.f, s2 = 0.f;
#pragma unroll
  for (int i = 0; i < 4; ++i) {
    v[i] = xr[i] + yr[i];
    s += v[i];
    s2 += v[i] * v[i];
  }
  red[tid] = s;
  red2[tid] = s2;
  __syncthreads();
  for (int st = 128; st > 0; st >>= 1) {
    if (tid < st) {
      red[tid] += red[tid + st];
      red2[tid] += red2[tid + st];
    }
    __syncthreads();
  }
  const float mu = red[0] / (float)D;
  const float var = red2[0] / (float)D - mu * mu;
  const float inv = rsqrtf(var + 1e-5f);
#pragma unroll
  for (int i = 0; i < 4; ++i) {
    const int c = tid * 4 + i;
    const float o = (v[i] - mu) * inv * g[c] + bta[c];
    outF[(long)row * D + c] = o;
    if (outB) outB[(long)row * D + c] = (bf16)o;
  }
}

// ---------------------------------------------------------------------------
// Host orchestration
// ---------------------------------------------------------------------------
extern "C" void kernel_launch(void* const* d_in, const int* in_sizes, int n_in,
                              void* d_out, int out_size, void* d_ws,
                              size_t ws_size, hipStream_t stream) {
  (void)in_sizes; (void)n_in; (void)out_size;
  const float* tgt    = (const float*)d_in[0];
  const float* memry  = (const float*)d_in[1];
  // d_in[2] = tgt_mask (causal; implemented analytically)
  const float* Wqkv = (const float*)d_in[3];
  const float* bqkv = (const float*)d_in[4];
  const float* Wo_sa = (const float*)d_in[5];
  const float* bo_sa = (const float*)d_in[6];
  const float* Wq = (const float*)d_in[7];
  const float* bq = (const float*)d_in[8];
  const float* Wk = (const float*)d_in[9];
  const float* bk = (const float*)d_in[10];
  const float* Wv = (const float*)d_in[11];
  const float* bv = (const float*)d_in[12];
  const float* Wo_cx = (const float*)d_in[13];
  const float* bo_cx = (const float*)d_in[14];
  const float* W1 = (const float*)d_in[15];
  const float* b1 = (const float*)d_in[16];
  const float* W2 = (const float*)d_in[17];
  const float* b2 = (const float*)d_in[18];
  const float* g_mha = (const float*)d_in[19];
  const float* bn_mha = (const float*)d_in[20];
  const float* g_crx = (const float*)d_in[21];
  const float* bn_crx = (const float*)d_in[22];
  const float* g_ffn = (const float*)d_in[23];
  const float* bn_ffn = (const float*)d_in[24];

  const long rows = (long)BBB * SS;  // 4096 query rows (== B*M too)

  char* ws = (char*)d_ws;
  size_t off = 0;
  auto balloc = [&](size_t bytes) -> char* {
    char* p = ws + off;
    off += (bytes + 255) & ~(size_t)255;
    return p;
  };

  bf16* WqkvT = (bf16*)balloc((size_t)DD * 3 * DD * 2);
  bf16* WosaT = (bf16*)balloc((size_t)DD * DD * 2);
  bf16* WqT   = (bf16*)balloc((size_t)DD * DD * 2);
  bf16* WkT   = (bf16*)balloc((size_t)DD * DD * 2);
  bf16* WvT   = (bf16*)balloc((size_t)DD * DD * 2);
  bf16* WocxT = (bf16*)balloc((size_t)DD * DD * 2);
  bf16* W1T   = (bf16*)balloc((size_t)DD * FFD * 2);
  bf16* W2T   = (bf16*)balloc((size_t)FFD * DD * 2);
  bf16* tgt_bf = (bf16*)balloc((size_t)rows * DD * 2);
  bf16* mem_bf = (bf16*)balloc((size_t)rows * DD * 2);
  float* qkv_f = (float*)balloc((size_t)rows * 3 * DD * 4);  // reused as q/k/v cx
  float* qcx = qkv_f;
  float* kcx = qkv_f + rows * DD;
  float* vcx = qkv_f + 2 * rows * DD;
  bf16* attn_bf = (bf16*)balloc((size_t)rows * DD * 2);      // attn1 then attn2
  float* branch = (float*)balloc((size_t)rows * DD * 4);     // sa, cx, ff2
  float* agg1 = (float*)balloc((size_t)rows * DD * 4);
  float* agg2 = (float*)balloc((size_t)rows * DD * 4);
  bf16* aggn_bf = (bf16*)balloc((size_t)rows * DD * 2);      // agg1_bf then agg2_bf
  bf16* ff1_bf = (bf16*)balloc((size_t)rows * FFD * 2);
  if (off > ws_size) return;

  auto nb = [](long n) { return (int)((n + 255) / 256); };

  // --- weight & activation conversion ---
  cvt_t_bf16_kernel<<<nb((long)DD * 3 * DD), 256, 0, stream>>>(Wqkv, WqkvT, DD, 3 * DD);
  cvt_t_bf16_kernel<<<nb((long)DD * DD), 256, 0, stream>>>(Wo_sa, WosaT, DD, DD);
  cvt_t_bf16_kernel<<<nb((long)DD * DD), 256, 0, stream>>>(Wq, WqT, DD, DD);
  cvt_t_bf16_kernel<<<nb((long)DD * DD), 256, 0, stream>>>(Wk, WkT, DD, DD);
  cvt_t_bf16_kernel<<<nb((long)DD * DD), 256, 0, stream>>>(Wv, WvT, DD, DD);
  cvt_t_bf16_kernel<<<nb((long)DD * DD), 256, 0, stream>>>(Wo_cx, WocxT, DD, DD);
  cvt_t_bf16_kernel<<<nb((long)DD * FFD), 256, 0, stream>>>(W1, W1T, DD, FFD);
  cvt_t_bf16_kernel<<<nb((long)FFD * DD), 256, 0, stream>>>(W2, W2T, FFD, DD);
  cvt_bf16_kernel<<<nb(rows * DD), 256, 0, stream>>>(tgt, tgt_bf, rows * DD);
  cvt_bf16_kernel<<<nb(rows * DD), 256, 0, stream>>>(memry, mem_bf, rows * DD);

  const dim3 blk(256);
  // --- self-attention sub-block ---
  gemm_bf16_kernel<<<dim3(3 * DD / 128, rows / 128), blk, 0, stream>>>(
      tgt_bf, WqkvT, bqkv, qkv_f, nullptr, (int)rows, 3 * DD, DD, 0);
  flash_attn_kernel<true><<<dim3(SS / 64, HHH, BBB), 128, 0, stream>>>(
      qkv_f, qkv_f + DD, qkv_f + 2 * DD, attn_bf, SS, SS,
      (long)SS * 3 * DD, 3 * DD, (long)SS * 3 * DD, 3 * DD,
      (long)SS * 3 * DD, 3 * DD, DD);
  gemm_bf16_kernel<<<dim3(DD / 128, rows / 128), blk, 0, stream>>>(
      attn_bf, WosaT, bo_sa, branch, nullptr, (int)rows, DD, DD, 0);
  add_ln_kernel<<<(int)rows, 256, 0, stream>>>(tgt, branch, g_mha, bn_mha, agg1,
                                               aggn_bf, DD);

  // --- cross-attention sub-block ---
  gemm_bf16_kernel<<<dim3(DD / 128, rows / 128), blk, 0, stream>>>(
      aggn_bf, WqT, bq, qcx, nullptr, (int)rows, DD, DD, 0);
  gemm_bf16_kernel<<<dim3(DD / 128, rows / 128), blk, 0, stream>>>(
      mem_bf, WkT, bk, kcx, nullptr, (int)rows, DD, DD, 0);
  gemm_bf16_kernel<<<dim3(DD / 128, rows / 128), blk, 0, stream>>>(
      mem_bf, WvT, bv, vcx, nullptr, (int)rows, DD, DD, 0);
  flash_attn_kernel<false><<<dim3(SS / 64, HHH, BBB), 128, 0, stream>>>(
      qcx, kcx, vcx, attn_bf, SS, MMK, (long)SS * DD, DD, (long)MMK * DD, DD,
      (long)MMK * DD, DD, DD);
  gemm_bf16_kernel<<<dim3(DD / 128, rows / 128), blk, 0, stream>>>(
      attn_bf, WocxT, bo_cx, branch, nullptr, (int)rows, DD, DD, 0);
  add_ln_kernel<<<(int)rows, 256, 0, stream>>>(agg1, branch, g_crx, bn_crx, agg2,
                                               aggn_bf, DD);

  // --- FFN sub-block ---
  gemm_bf16_kernel<<<dim3(FFD / 128, rows / 128), blk, 0, stream>>>(
      aggn_bf, W1T, b1, nullptr, ff1_bf, (int)rows, FFD, DD, 1);
  gemm_bf16_kernel<<<dim3(DD / 128, rows / 128), blk, 0, stream>>>(
      ff1_bf, W2T, b2, branch, nullptr, (int)rows, DD, FFD, 0);
  add_ln_kernel<<<(int)rows, 256, 0, stream>>>(agg2, branch, g_ffn, bn_ffn,
                                               (float*)d_out, nullptr, DD);
}
